// GNN_70214125355382
// MI455X (gfx1250) — matrix operand order repaired and checked by reading, not compiled, not run on hardware
//
#include <hip/hip_runtime.h>

#define NN 100000
#define EE 1600000
#define GG 512
#define CIN 128
#define HDIM 256
#define COUT 10
#define BN_EPS 1e-5f

typedef __attribute__((ext_vector_type(16))) __bf16 v16bf;
typedef __attribute__((ext_vector_type(8)))  __bf16 v8bf;
typedef __attribute__((ext_vector_type(8)))  float  v8f;
typedef __attribute__((ext_vector_type(4)))  unsigned int v4u;
typedef __attribute__((ext_vector_type(8)))  unsigned int v8u;

// ---------------- Tensor Data Mover: contiguous global -> LDS copy ----------------
// Builds a D# (CDNA5 ISA): group0 = {count=1, lds_addr, global_addr, type=2},
// group1 = {data_size=8B, tensor_dim0=n8, tensor_dim1=1, tile_dim0=n8, tile_dim1=1, stride0=n8}.
// 2-operand form leaves VADDR2/VADDR3 = NULL (<=2D tensor). Portable inline asm (bypasses the
// builtin whose arity differs between toolchains).
__device__ __forceinline__ void tdm_load_to_lds(const void* gptr, unsigned lds_off, unsigned n8) {
  unsigned long long ga = (unsigned long long)(size_t)gptr;
  v4u g0;
  g0[0] = 1u;                                            // count=1, user mode
  g0[1] = lds_off;                                       // lds_addr (bytes)
  g0[2] = (unsigned)(ga & 0xffffffffu);                  // global_addr[31:0]
  g0[3] = (unsigned)((ga >> 32) & 0x01ffffffu) | (2u << 30); // global_addr[56:32] | type=2
  v8u g1;
  g1[0] = 3u << 16;                                      // data_size = 3 (8 bytes)
  g1[1] = (n8 & 0xffffu) << 16;                          // tensor_dim0[15:0] @ bits 63:48
  g1[2] = ((n8 >> 16) & 0xffffu) | (1u << 16);           // tensor_dim0[31:16] | tensor_dim1=1
  g1[3] = (n8 & 0xffffu) << 16;                          // tile_dim0 @ bits 127:112
  g1[4] = 1u;                                            // tile_dim1 = 1
  g1[5] = n8;                                            // tensor_dim0_stride[31:0]
  g1[6] = 0u;
  g1[7] = 0u;
  asm volatile("tensor_load_to_lds %0, %1" :: "s"(g0), "s"(g1) : "memory");
}

// ---------------- elementwise helpers ----------------
__global__ void k_copy_f32x4(const float* __restrict__ s, float* __restrict__ d, int n4) {
  int i = blockIdx.x * blockDim.x + threadIdx.x;
  int st = gridDim.x * blockDim.x;
  const float4* sp = (const float4*)s;
  float4* dp = (float4*)d;
  for (; i < n4; i += st) dp[i] = sp[i];
}

__global__ void k_zero_f32(float* __restrict__ d, int n) {
  int i = blockIdx.x * blockDim.x + threadIdx.x;
  int st = gridDim.x * blockDim.x;
  for (; i < n; i += st) d[i] = 0.0f;
}

__global__ void k_f32_to_bf16(const float* __restrict__ s, __bf16* __restrict__ d, int n) {
  int i = blockIdx.x * blockDim.x + threadIdx.x;
  int st = gridDim.x * blockDim.x;
  for (; i < n; i += st) d[i] = (__bf16)s[i];
}

// ---------------- edge aggregation: agg[dst] += x[src] ----------------
__global__ void k_edge_agg(const float* __restrict__ x, const int* __restrict__ esrc,
                           const int* __restrict__ edst, float* __restrict__ agg, int cq) {
  const int C = cq * 4;
  int i = blockIdx.x * blockDim.x + threadIdx.x;
  int st = gridDim.x * blockDim.x;
  const int total = EE * cq;
  for (; i < total; i += st) {
    int e = i / cq;
    int c = (i - e * cq) * 4;
    const float4 v = *(const float4*)(x + (size_t)esrc[e] * C + c);
    float* p = agg + (size_t)edst[e] * C + c;
    atomicAdd(p + 0, v.x);
    atomicAdd(p + 1, v.y);
    atomicAdd(p + 2, v.z);
    atomicAdd(p + 3, v.w);
  }
}

// ---------------- pack weight W[K x 256] f32 -> B-fragment bf16 layout ----------------
// out index = (((kb*16 + nt)*32 + lane)*16 + j2); element = W[(kb*32 + (lane<16?0:8) + (j2&7) + ((j2&8)<<1))*256 + nt*16 + (lane&15)]
__global__ void k_pack_w(const float* __restrict__ W, __bf16* __restrict__ out, int K) {
  int i = blockIdx.x * blockDim.x + threadIdx.x;
  int st = gridDim.x * blockDim.x;
  const int total = K * HDIM;
  for (; i < total; i += st) {
    int j2 = i & 15;
    int lane = (i >> 4) & 31;
    int frag = i >> 9;
    int nt = frag & 15;
    int kb = frag >> 4;
    int k = kb * 32 + ((lane < 16) ? 0 : 8) + (j2 & 7) + ((j2 & 8) << 1);
    int n = (nt << 4) + (lane & 15);
    out[i] = (__bf16)W[(size_t)k * HDIM + n];
  }
}

// ---------------- WMMA GEMM: [M x K]bf16 @ [K x 256]bf16(packed, TDM-staged in LDS) ----
// One wave computes a 16x64 output slab (4 accumulators) to reuse each A fragment 4x.
// B (K*256 bf16) is DMA'd into LDS once per block by the Tensor Data Mover.
// Per k-step: issue all 8 ds_load_b128 for the 4 B fragments, then 4 back-to-back WMMAs
// (disjoint accumulators -> no D->A/B hazard), hiding LDS latency behind the XDL pipe.
template<int K, bool FUSE_BN>
__global__ void k_gemm_lds(const __bf16* __restrict__ A, const __bf16* __restrict__ Bp,
                           const float* __restrict__ bias,
                           const float* __restrict__ gamma, const float* __restrict__ beta,
                           const float* __restrict__ mean,  const float* __restrict__ var,
                           __bf16* __restrict__ out_bf, float* __restrict__ out_f) {
  extern __shared__ __bf16 sB[];  // K*256 bf16 (64KB or 128KB) of 320KB WGP LDS

  if (threadIdx.x < 32) {  // wave 0 issues the DMA and waits on TENSORcnt
    tdm_load_to_lds(Bp, __builtin_amdgcn_groupstaticsize(), (unsigned)(K * HDIM / 4));
    __builtin_amdgcn_s_wait_tensorcnt(0);
  }
  __syncthreads();

  const int lane = threadIdx.x & 31;
  const int wid  = blockIdx.x * 8 + (threadIdx.x >> 5);
  const int tm = wid >> 2;          // 6250 M-tiles
  const int ng = wid & 3;           // 4 n-groups of 64 columns
  if (tm >= NN / 16) return;        // wave-uniform (exact tiling)

  const int row = tm * 16 + (lane & 15);
  const int kb1 = (lane < 16) ? 0 : 8;   // A-fragment K base per ISA layout
  const __bf16* arow = A + (size_t)row * K + kb1;

  v8f acc[4] = {{}, {}, {}, {}};
#pragma unroll
  for (int kb = 0; kb < K / 32; ++kb) {
    v8bf alo = *(const v8bf*)(arow + kb * 32);        // K = kb*32 + kb1 + [0..7]
    v8bf ahi = *(const v8bf*)(arow + kb * 32 + 16);   // K = kb*32 + kb1 + [16..23]
    v16bf a = __builtin_shufflevector(alo, ahi, 0,1,2,3,4,5,6,7,8,9,10,11,12,13,14,15);
    const __bf16* bbase = sB + ((size_t)(kb * 16 + ng * 4) * 32 + lane) * 16;
    v16bf b0 = *(const v16bf*)(bbase + 0 * 512);      // 512 halfs per n-tile fragment
    v16bf b1 = *(const v16bf*)(bbase + 1 * 512);
    v16bf b2 = *(const v16bf*)(bbase + 2 * 512);
    v16bf b3 = *(const v16bf*)(bbase + 3 * 512);
    acc[0] = __builtin_amdgcn_wmma_f32_16x16x32_bf16(false, a, false, b0, (short)0, acc[0], false, false);
    acc[1] = __builtin_amdgcn_wmma_f32_16x16x32_bf16(false, a, false, b1, (short)0, acc[1], false, false);
    acc[2] = __builtin_amdgcn_wmma_f32_16x16x32_bf16(false, a, false, b2, (short)0, acc[2], false, false);
    acc[3] = __builtin_amdgcn_wmma_f32_16x16x32_bf16(false, a, false, b3, (short)0, acc[3], false, false);
  }

  const int rbase = tm * 16 + ((lane < 16) ? 0 : 8);
#pragma unroll
  for (int j = 0; j < 4; ++j) {
    const int col = (ng * 4 + j) * 16 + (lane & 15);
    const float bv = bias[col];
    if (FUSE_BN) {
      const float sc = gamma[col] * __frsqrt_rn(var[col] + BN_EPS);
      const float sh = beta[col] - mean[col] * sc;
#pragma unroll
      for (int r = 0; r < 8; ++r) {
        float v = (acc[j][r] + bv) * sc + sh;
        out_f[(size_t)(rbase + r) * HDIM + col] = fmaxf(v, 0.0f);
      }
    } else {
#pragma unroll
      for (int r = 0; r < 8; ++r) {
        float v = fmaxf(acc[j][r] + bv, 0.0f);
        out_bf[(size_t)(rbase + r) * HDIM + col] = (__bf16)v;
      }
    }
  }
}

// ---------------- pooling + classifier ----------------
__global__ void k_pool_sum(const float* __restrict__ h, const int* __restrict__ batch,
                           float* __restrict__ pool) {
  int i = blockIdx.x * blockDim.x + threadIdx.x;
  int st = gridDim.x * blockDim.x;
  const int total = NN * HDIM;
  for (; i < total; i += st) {
    int node = i >> 8;
    int c = i & 255;
    atomicAdd(pool + (size_t)batch[node] * HDIM + c, h[i]);
  }
}

__global__ void k_pool_cnt(const int* __restrict__ batch, float* __restrict__ cnt) {
  int i = blockIdx.x * blockDim.x + threadIdx.x;
  if (i < NN) atomicAdd(cnt + batch[i], 1.0f);
}

__global__ void k_classifier(const float* __restrict__ pool, const float* __restrict__ cnt,
                             const float* __restrict__ Wc, const float* __restrict__ bc,
                             float* __restrict__ out) {
  int i = blockIdx.x * blockDim.x + threadIdx.x;
  if (i >= GG * COUT) return;
  int g = i / COUT;
  int o = i - g * COUT;
  float inv = 1.0f / fmaxf(cnt[g], 1.0f);
  float s = 0.0f;
  for (int k = 0; k < HDIM; ++k) s += pool[(size_t)g * HDIM + k] * Wc[k * COUT + o];
  out[i] = s * inv + bc[o];
}

extern "C" void kernel_launch(void* const* d_in, const int* in_sizes, int n_in,
                              void* d_out, int out_size, void* d_ws, size_t ws_size,
                              hipStream_t stream) {
  (void)in_sizes; (void)n_in; (void)out_size; (void)ws_size;
  const float* x     = (const float*)d_in[0];
  const int*   esrc  = (const int*)d_in[1];
  const int*   edst  = (const int*)d_in[2];
  const int*   batch = (const int*)d_in[3];
  const float* W1_0  = (const float*)d_in[4];
  const float* b1_0  = (const float*)d_in[5];
  const float* W2_0  = (const float*)d_in[6];
  const float* b2_0  = (const float*)d_in[7];
  const float* W1_r  = (const float*)d_in[8];
  const float* b1_r  = (const float*)d_in[9];
  const float* W2_r  = (const float*)d_in[10];
  const float* b2_r  = (const float*)d_in[11];
  const float* bn_g  = (const float*)d_in[12];
  const float* bn_b  = (const float*)d_in[13];
  const float* bn_m  = (const float*)d_in[14];
  const float* bn_v  = (const float*)d_in[15];
  const float* Wc    = (const float*)d_in[16];
  const float* bc    = (const float*)d_in[17];
  float* out = (float*)d_out;

  char* ws = (char*)d_ws;
  size_t off = 0;
  auto alloc = [&](size_t bytes) -> char* {
    off = (off + 255) & ~(size_t)255;
    char* p = ws + off;
    off += bytes;
    return p;
  };

  float*  f0   = (float*)alloc((size_t)NN * HDIM * 4);  // ping
  float*  f1   = (float*)alloc((size_t)NN * HDIM * 4);  // pong
  __bf16* Abf  = (__bf16*)alloc((size_t)NN * HDIM * 2);
  __bf16* mid  = (__bf16*)alloc((size_t)NN * HDIM * 2);
  __bf16* w1p  = (__bf16*)alloc((size_t)HDIM * HDIM * 2);
  __bf16* w2p  = (__bf16*)alloc((size_t)HDIM * HDIM * 2);
  float*  pool = (float*)alloc((size_t)GG * HDIM * 4 + GG * 4);
  float*  cnt  = pool + (size_t)GG * HDIM;

  // 16x64 slab per wave: 6250 m-tiles * 4 n-groups = 25000 waves; 8 waves / block
  const int gemm_blocks = (NN / 16) * 4 / 8;
  const float* cur = x;
  int C = CIN;
  float* pingpong[2] = { f0, f1 };

  for (int l = 0; l < 4; ++l) {
    const float* W1 = (l == 0) ? W1_0 : W1_r + (size_t)(l - 1) * HDIM * HDIM;
    const float* b1 = (l == 0) ? b1_0 : b1_r + (size_t)(l - 1) * HDIM;
    const float* W2 = (l == 0) ? W2_0 : W2_r + (size_t)(l - 1) * HDIM * HDIM;
    const float* b2 = (l == 0) ? b2_0 : b2_r + (size_t)(l - 1) * HDIM;
    float* agg = pingpong[l & 1];
    const int K = C;

    // pack weights into WMMA B-fragment order
    k_pack_w<<<(K * HDIM) / 256, 256, 0, stream>>>(W1, w1p, K);
    k_pack_w<<<(HDIM * HDIM) / 256, 256, 0, stream>>>(W2, w2p, HDIM);

    // agg = x ; agg[dst] += x[src]
    k_copy_f32x4<<<2048, 256, 0, stream>>>(cur, agg, NN * C / 4);
    k_edge_agg<<<8192, 256, 0, stream>>>(cur, esrc, edst, agg, C / 4);

    // bf16 conversion of (x + agg)
    k_f32_to_bf16<<<4096, 256, 0, stream>>>(agg, Abf, NN * C);

    // GEMM1: relu(A @ W1 + b1) -> bf16 (B staged to LDS via TDM: 64KB or 128KB dynamic LDS)
    if (l == 0)
      k_gemm_lds<128, false><<<gemm_blocks, 256, 128 * HDIM * 2, stream>>>(Abf, w1p, b1,
          nullptr, nullptr, nullptr, nullptr, mid, nullptr);
    else
      k_gemm_lds<256, false><<<gemm_blocks, 256, 256 * HDIM * 2, stream>>>(Abf, w1p, b1,
          nullptr, nullptr, nullptr, nullptr, mid, nullptr);

    // GEMM2: relu(BN(mid @ W2 + b2)) -> f32 (reuse agg buffer; GEMM2 reads only `mid`)
    k_gemm_lds<256, true><<<gemm_blocks, 256, 256 * HDIM * 2, stream>>>(mid, w2p, b2,
        bn_g + (size_t)l * HDIM, bn_b + (size_t)l * HDIM,
        bn_m + (size_t)l * HDIM, bn_v + (size_t)l * HDIM, nullptr, agg);

    cur = agg;
    C = HDIM;
  }

  // global mean pool + classifier
  k_zero_f32<<<(GG * HDIM + GG + 255) / 256, 256, 0, stream>>>(pool, GG * HDIM + GG);
  k_pool_sum<<<8192, 256, 0, stream>>>(pingpong[1], batch, pool);  // layer 3 output lives in f1
  k_pool_cnt<<<(NN + 255) / 256, 256, 0, stream>>>(batch, cnt);
  k_classifier<<<(GG * COUT + 255) / 256, 256, 0, stream>>>(pool, cnt, Wc, bc, out);
}
